// CoKT_78142634983606
// MI455X (gfx1250) — compile-verified, matrix operand order checked by reading, not run on hardware
//
#include <hip/hip_runtime.h>
#include <hip/hip_bf16.h>
#include <math.h>

typedef __bf16 bf16;
typedef __attribute__((ext_vector_type(16))) __bf16 v16bf;
typedef __attribute__((ext_vector_type(8)))  __bf16 v8bf;
typedef __attribute__((ext_vector_type(4)))  __bf16 v4bf;
typedef __attribute__((ext_vector_type(8)))  float  v8f;

#define DD   64
#define TT   128
#define BB   64
#define RR   20
#define SEEN 10
#define HH   4
#define CC   4

#define BLK_B    16
#define NTHREADS 256
#define NWAVES   8

// ---------------- LDS arena (bytes) ----------------
#define OFF_H      0u            // 16*64  f32  = 4096
#define OFF_HB     4096u         // 16*64  bf16 = 2048
#define OFF_V      6144u         // 16*128 f32  = 8192
#define OFF_RBUF   14336u        // 16*129 f32  (pad 8320)
#define OFF_BGR    22656u        // 160 f32     = 640
#define OFF_ACAT   23296u        // 16*640 bf16 = 20480
#define OFF_QB     43776u        // 16*256 bf16 = 8192
#define OFF_BGA    51968u        // 16*384 bf16 = 12288
#define OFF_HISA   64256u        // 16*128 bf16 = 4096
#define OFF_SC     68352u        // 16*4*10 f32 = 2560
#define OFF_WR     70912u        // 160 f32     = 640
#define OFF_WNR    71552u        // 160 f32     = 640
#define OFF_MISC   72192u        // dw0,dw1,hm  = 64
#define OFF_BCOMB  72256u        // 384 f32     = 1536
// R1: allhv bf16 (160x192=61440)  |  { grub bf16 16x256 (8192), g1 f32 16x192 (12288), g2 f32 16x192 (12288) }
#define OFF_R1     73792u
// R2: p f32 16x192 | hsc f32 16x129 | peer f32 16x384 (=24576)
#define OFF_R2     135232u
#define SMEM_TOTAL 159808u

// ---------------- packed-weight region in d_ws ----------------
// ws layout: hbuf [B][129][64] f32 | vbuf [B][129][128] f32 | packed bf16 weights
#define WS_HBUF_ELEMS   ((long)BB * (TT + 1) * DD)          // 528384
#define WS_VBUF_ELEMS   ((long)BB * (TT + 1) * 2 * DD)      // 1056768
#define WS_PACK_BYTE    ((WS_HBUF_ELEMS + WS_VBUF_ELEMS) * 4) // 6340608, 32B aligned
// packed offsets (bf16 elements); each tile = 32 lanes * 16 bf16 = 512
#define OFFP_BATCH   0L        // 4 * (6kt*4nt*512)  = 49152
#define OFFP_BG      49152L    // 4 * (2kt*12nt*512) = 49152
#define OFFP_BGINT   98304L    // 12kt*24nt*512      = 147456
#define OFFP_HISINT  245760L   // 4kt*24nt*512       = 49152
#define OFFP_INMAP   294912L   // 20kt*16nt*512      = 163840
#define OFFP_IH      458752L   // 8kt*12nt*512       = 49152
#define OFFP_HH      507904L   // 2kt*12nt*512       = 12288

__device__ __forceinline__ int laneid() { return (int)(threadIdx.x & 31u); }

// ------------- weight pack: fp32 (strided) -> bf16 fragment-native -------------
// load side reads: Wp[ (((nt*nKt + kt)*32 + lane) << 4) + i ]  where
//   n = nt*16 + (lane&15);  k = kt*32 + ((lane>>4)<<3) + (i<8 ? i : i+8)
__global__ void pack_weight(const float* __restrict__ src, bf16* __restrict__ dst,
                            int K, int N, int ks, int ns, int nh, long src_h) {
    const int nKt = K >> 5, nNt = N >> 4;
    const long total = (long)nh * nKt * nNt * 512;
    for (long idx = (long)blockIdx.x * blockDim.x + threadIdx.x; idx < total;
         idx += (long)gridDim.x * blockDim.x) {
        int i  = (int)(idx & 15);
        int ln = (int)((idx >> 4) & 31);
        long tl = idx >> 9;
        int kt = (int)(tl % nKt);
        long tmp = tl / nKt;
        int nt = (int)(tmp % nNt);
        int h  = (int)(tmp / nNt);
        int n = nt * 16 + (ln & 15);
        int k = kt * 32 + ((ln >> 4) << 3) + (i < 8 ? i : i + 8);
        dst[idx] = (bf16)src[h * src_h + (long)k * ks + (long)n * ns];
    }
}

// ------------- WMMA fragment helpers -------------
// A fragment: two contiguous 16B runs per lane -> 2x ds_load_b128
__device__ __forceinline__ v16bf load_a_frag(const bf16* A, int lda, int k0) {
    const int ln = laneid();
    const bf16* p = A + (ln & 15) * lda + k0 + ((ln >> 4) << 3);
    v8bf lo = *(const v8bf*)p;
    v8bf hi = *(const v8bf*)(p + 16);
    return __builtin_shufflevector(lo, hi, 0, 1, 2, 3, 4, 5, 6, 7,
                                          8, 9, 10, 11, 12, 13, 14, 15);
}

__device__ __forceinline__ v8f tile_mm_pk(const bf16* A, int lda, const bf16* Wp,
                                          int nKt, int nt, v8f acc) {
    const int ln = laneid();
    for (int kt = 0; kt < nKt; ++kt) {
        v16bf a = load_a_frag(A, lda, kt << 5);
        v16bf b = *(const v16bf*)(Wp + (((long)(nt * nKt + kt) * 32 + ln) << 4)); // 32B load
        acc = __builtin_amdgcn_wmma_f32_16x16x32_bf16(false, a, false, b, (short)0, acc, false, false);
    }
    return acc;
}

__device__ __forceinline__ void store_c_f32(float* C, int ldc, int n0, v8f acc, const float* bias) {
    const int ln = laneid();
    const int n  = n0 + (ln & 15);
    const int mb = (ln >> 4) << 3;
    const float bv = bias ? bias[n] : 0.0f;
#pragma unroll
    for (int i = 0; i < 8; ++i) C[(mb + i) * ldc + n] = acc[i] + bv;
}

__device__ __forceinline__ void store_c_bf16(bf16* C, int ldc, int n0, v8f acc, const float* bias) {
    const int ln = laneid();
    const int n  = n0 + (ln & 15);
    const int mb = (ln >> 4) << 3;
    const float bv = bias ? bias[n] : 0.0f;
#pragma unroll
    for (int i = 0; i < 8; ++i) C[(mb + i) * ldc + n] = (bf16)(acc[i] + bv);
}

// ---------------- main persistent kernel ----------------
__global__ __launch_bounds__(NTHREADS, 1)
void cokt_persistent(const int* __restrict__ prob_id, const int* __restrict__ skills,
                     const int* __restrict__ response, const int* __restrict__ bg_index,
                     const int* __restrict__ mem_prob_ids, const int* __restrict__ mem_resp,
                     const int* __restrict__ mem_concepts,
                     const float* __restrict__ states_mem, const float* __restrict__ concept_emb,
                     const float* __restrict__ prob_emb,
                     const float* __restrict__ W_pred, const float* __restrict__ b_pred,
                     const float* __restrict__ b_ih, const float* __restrict__ b_hh,
                     const float* __restrict__ b_batch,
                     const float* __restrict__ b_bgint, const float* __restrict__ b_hisint,
                     const float* __restrict__ b_inmap,
                     const float* __restrict__ dir_w, const float* __restrict__ heads_map,
                     float* __restrict__ out, float* __restrict__ ws) {
    extern __shared__ __align__(16) unsigned char smem[];
    float* sH    = (float*)(smem + OFF_H);
    bf16*  sHB   = (bf16*) (smem + OFF_HB);
    float* sV    = (float*)(smem + OFF_V);
    float* sRB   = (float*)(smem + OFF_RBUF);
    float* sBGR  = (float*)(smem + OFF_BGR);
    bf16*  sACAT = (bf16*) (smem + OFF_ACAT);
    bf16*  sQB   = (bf16*) (smem + OFF_QB);
    bf16*  sBGA  = (bf16*) (smem + OFF_BGA);
    bf16*  sHISA = (bf16*) (smem + OFF_HISA);
    float* sSC   = (float*)(smem + OFF_SC);
    float* sWR   = (float*)(smem + OFF_WR);
    float* sWNR  = (float*)(smem + OFF_WNR);
    float* sMISC = (float*)(smem + OFF_MISC);
    float* sBC   = (float*)(smem + OFF_BCOMB);
    bf16*  sALLHV= (bf16*) (smem + OFF_R1);               // [160][192]
    bf16*  sGRUB = (bf16*) (smem + OFF_R1);               // [16][256]
    float* sG1   = (float*)(smem + OFF_R1 + 8192u);       // [16][192]
    float* sG2   = (float*)(smem + OFF_R1 + 20480u);      // [16][192]
    float* sP    = (float*)(smem + OFF_R2);               // [16][192]
    float* sHSC  = (float*)(smem + OFF_R2);               // [16][129]
    float* sPEER = (float*)(smem + OFF_R2);               // [16][384]

    const int tid = (int)threadIdx.x;
    const int wv  = tid >> 5;
    const int b0  = (int)blockIdx.x * BLK_B;

    float* hbuf = ws;
    float* vbuf = ws + WS_HBUF_ELEMS;
    const bf16* wp = (const bf16*)((const char*)ws + WS_PACK_BYTE);
    const bf16* wpBATCH  = wp + OFFP_BATCH;
    const bf16* wpBG     = wp + OFFP_BG;
    const bf16* wpBGINT  = wp + OFFP_BGINT;
    const bf16* wpHISINT = wp + OFFP_HISINT;
    const bf16* wpINMAP  = wp + OFFP_INMAP;
    const bf16* wpIH     = wp + OFFP_IH;
    const bf16* wpHH     = wp + OFFP_HH;

    // ---- init ----
    const float e0 = __expf(dir_w[0]), e1 = __expf(dir_w[1]);
    const float dw0 = e0 / (e0 + e1), dw1 = e1 / (e0 + e1);
    if (tid == 0) { sMISC[0] = dw0; sMISC[1] = dw1; }
    if (tid < HH) sMISC[2 + tid] = heads_map[tid];
    for (int i = tid; i < 6 * DD; i += NTHREADS) sBC[i] = dw0 * b_bgint[i] + dw1 * b_hisint[i];
    for (int i = tid; i < BLK_B * DD; i += NTHREADS) sH[i] = 0.f;
    if (tid < BLK_B) sRB[tid * (TT + 1)] = 0.f;
    for (int i = tid; i < BLK_B * DD; i += NTHREADS) {
        int bl = i / DD, d = i % DD;
        hbuf[((long)(b0 + bl) * (TT + 1)) * DD + d] = 0.f;
    }
    for (int i = tid; i < BLK_B * 2 * DD; i += NTHREADS) {
        int bl = i / (2 * DD), c = i % (2 * DD);
        vbuf[((long)(b0 + bl) * (TT + 1)) * (2 * DD) + c] = 0.f;
    }
    __syncthreads();

    for (int t = 0; t < TT; ++t) {
        // ========== Phase A: gathers -> v, hidden_v(acat), all_hv, bg_r (float4 vectorized) ==========
        for (int i = tid; i < BLK_B * 32; i += NTHREADS) {   // v: 16 rows x 32 groups of 4
            int bl = i >> 5, c4 = (i & 31) << 2;
            int b = b0 + bl;
            float v0, v1, v2, v3;
            if (c4 < DD) {
                float s0 = 0.f, s1 = 0.f, s2 = 0.f, s3 = 0.f; int cnt = 0;
#pragma unroll
                for (int j = 0; j < CC; ++j) {
                    int sk = skills[(t * BB + b) * CC + j];
                    if (sk != 0) {
                        const float4 e = *(const float4*)(concept_emb + (long)(sk - 1) * DD + c4);
                        s0 += e.x; s1 += e.y; s2 += e.z; s3 += e.w; cnt++;
                    }
                }
                float inv = 1.f / (float)(cnt == 0 ? 1 : cnt);
                v0 = s0 * inv; v1 = s1 * inv; v2 = s2 * inv; v3 = s3 * inv;
            } else {
                int pid = prob_id[t * BB + b];
                if (pid != 0) {
                    const float4 e = *(const float4*)(prob_emb + (long)(pid - 1) * DD + (c4 - DD));
                    v0 = e.x; v1 = e.y; v2 = e.z; v3 = e.w;
                } else v0 = v1 = v2 = v3 = 0.f;
            }
            float4 o; o.x = v0; o.y = v1; o.z = v2; o.w = v3;
            *(float4*)(sV + bl * 2 * DD + c4) = o;
            v4bf ob; ob[0] = (bf16)v0; ob[1] = (bf16)v1; ob[2] = (bf16)v2; ob[3] = (bf16)v3;
            *(v4bf*)(sACAT + bl * 640 + DD + c4) = ob;
        }
        for (int i = tid; i < BLK_B * 16; i += NTHREADS) {   // acat[0:64] = h
            int bl = i >> 4, d4 = (i & 15) << 2;
            const float* hp = sH + bl * DD + d4;
            v4bf ob; ob[0] = (bf16)hp[0]; ob[1] = (bf16)hp[1]; ob[2] = (bf16)hp[2]; ob[3] = (bf16)hp[3];
            *(v4bf*)(sACAT + bl * 640 + d4) = ob;
        }
        for (int i = tid; i < 160 * 48; i += NTHREADS) {     // all_hv: 160 rows x 48 groups of 4
            int row = i / 48, c4 = (i % 48) << 2;
            int bl = row / SEEN, s = row % SEEN;
            int gi = bg_index[((long)t * BB + (b0 + bl)) * RR + s];
            float v0, v1, v2, v3;
            if (c4 < DD) {
                const float4 e = *(const float4*)(states_mem + (long)gi * DD + c4);
                v0 = e.x; v1 = e.y; v2 = e.z; v3 = e.w;
            } else if (c4 < 2 * DD) {
                float s0 = 0.f, s1 = 0.f, s2 = 0.f, s3 = 0.f; int cnt = 0;
#pragma unroll
                for (int j = 0; j < CC; ++j) {
                    int sk = mem_concepts[(long)gi * CC + j];
                    if (sk != 0) {
                        const float4 e = *(const float4*)(concept_emb + (long)(sk - 1) * DD + (c4 - DD));
                        s0 += e.x; s1 += e.y; s2 += e.z; s3 += e.w; cnt++;
                    }
                }
                float inv = 1.f / (float)(cnt == 0 ? 1 : cnt);
                v0 = s0 * inv; v1 = s1 * inv; v2 = s2 * inv; v3 = s3 * inv;
            } else {
                int pid = mem_prob_ids[gi];
                if (pid != 0) {
                    const float4 e = *(const float4*)(prob_emb + (long)(pid - 1) * DD + (c4 - 2 * DD));
                    v0 = e.x; v1 = e.y; v2 = e.z; v3 = e.w;
                } else v0 = v1 = v2 = v3 = 0.f;
            }
            v4bf ob; ob[0] = (bf16)v0; ob[1] = (bf16)v1; ob[2] = (bf16)v2; ob[3] = (bf16)v3;
            *(v4bf*)(sALLHV + row * 192 + c4) = ob;
        }
        for (int i = tid; i < 160; i += NTHREADS) {
            int bl = i / SEEN, s = i % SEEN;
            int gi = bg_index[((long)t * BB + (b0 + bl)) * RR + s];
            sBGR[i] = (float)mem_resp[gi];
        }
        __syncthreads();

        // ========== Phase B: q = hidden_v @ W_batch + b_batch -> bf16 directly ==========
        for (int j = wv; j < 16; j += NWAVES) {
            int h = j >> 2, nt = j & 3;
            v8f acc;
#pragma unroll
            for (int i = 0; i < 8; ++i) acc[i] = 0.f;
            acc = tile_mm_pk(sACAT, 640, wpBATCH + (long)h * 12288, 6, nt, acc);
            store_c_bf16(sQB, 256, h * DD + nt * 16, acc, b_batch);
        }
        __syncthreads();

        // ========== Phase C: p_h = q_h @ W_bg[h]^T; scores; softmax; bg_atts ==========
        const float scl_bg = 0.0721687836f; // 1/sqrt(3D)
        for (int h = 0; h < HH; ++h) {
            for (int j = wv; j < 12; j += NWAVES) {
                v8f acc;
#pragma unroll
                for (int i = 0; i < 8; ++i) acc[i] = 0.f;
                acc = tile_mm_pk(sQB + h * DD, 256, wpBG + (long)h * 12288, 2, j, acc);
                store_c_f32(sP, 192, j * 16, acc, nullptr);
            }
            __syncthreads();
            for (int i = tid; i < 160; i += NTHREADS) {
                int bl = i / SEEN, s = i % SEEN;
                const bf16* ar = sALLHV + (bl * SEEN + s) * 192;
                const float* pr = sP + bl * 192;
                float sum = 0.f;
                for (int k = 0; k < 192; ++k) sum += (float)ar[k] * pr[k];
                sSC[(bl * HH + h) * SEEN + s] = sum * scl_bg;
            }
            __syncthreads();
        }
        if (tid < BLK_B * HH) {
            float* sc = sSC + tid * SEEN;
            float m = sc[0];
            for (int s = 1; s < SEEN; ++s) m = fmaxf(m, sc[s]);
            float den = 0.f;
            for (int s = 0; s < SEEN; ++s) { sc[s] = __expf(sc[s] - m); den += sc[s]; }
            float inv = 1.f / den;
            for (int s = 0; s < SEEN; ++s) sc[s] *= inv;
        }
        __syncthreads();
        for (int i = tid; i < 160; i += NTHREADS) {
            int bl = i / SEEN, s = i % SEEN;
            float w = 0.f;
#pragma unroll
            for (int h = 0; h < HH; ++h) w += sMISC[2 + h] * sSC[(bl * HH + h) * SEEN + s];
            float r = sBGR[i];
            sWR[i] = w * r; sWNR[i] = w * (1.f - r);
        }
        __syncthreads();
        const float ldw0 = sMISC[0], ldw1 = sMISC[1];
        for (int i = tid; i < BLK_B * 192; i += NTHREADS) {
            int bl = i / 192, c = i % 192;
            float a1 = 0.f, a2 = 0.f;
#pragma unroll
            for (int s = 0; s < SEEN; ++s) {
                float x = (float)sALLHV[(bl * SEEN + s) * 192 + c];
                a1 += sWR[bl * SEEN + s] * x;
                a2 += sWNR[bl * SEEN + s] * x;
            }
            sBGA[bl * 384 + c]       = (bf16)(ldw0 * a1);
            sBGA[bl * 384 + 192 + c] = (bf16)(ldw0 * a2);
        }
        __syncthreads();

        // ========== Phase D: history attention ==========
        const int TL = t + 1;
        const float scl_his = 0.0883883476f; // 1/sqrt(2D)
        for (int i = tid; i < BLK_B * TL; i += NTHREADS) {
            int tp = i >> 4, bl = i & 15;
            const float* vb = vbuf + ((long)(b0 + bl) * (TT + 1) + tp) * (2 * DD);
            const float* vv = sV + bl * 2 * DD;
            float sum = 0.f;
            for (int k = 0; k < 2 * DD; k += 4) {
                float4 a = *(const float4*)(vv + k);
                float4 g = *(const float4*)(vb + k);
                sum += a.x * g.x + a.y * g.y + a.z * g.z + a.w * g.w;
            }
            sHSC[bl * (TT + 1) + tp] = sum * scl_his;
        }
        __syncthreads();
        if (tid < BLK_B) {
            float* sc = sHSC + tid * (TT + 1);
            float m = sc[0];
            for (int s = 1; s < TL; ++s) m = fmaxf(m, sc[s]);
            float den = 0.f;
            for (int s = 0; s < TL; ++s) { sc[s] = __expf(sc[s] - m); den += sc[s]; }
            float inv = 1.f / den;
            for (int s = 0; s < TL; ++s) sc[s] *= inv;
        }
        __syncthreads();
        for (int i = tid; i < BLK_B * 2 * DD; i += NTHREADS) {
            int bl = i >> 7, c = i & 127;
            int cc = c & 63; const bool top = c < DD;
            const float* hb = hbuf + ((long)(b0 + bl) * (TT + 1)) * DD + cc;
            float sum = 0.f;
            for (int s = 0; s < TL; ++s) {
                float w = sHSC[bl * (TT + 1) + s];
                float rb = sRB[bl * (TT + 1) + s];
                sum += w * (top ? rb : (1.f - rb)) * hb[(long)s * DD];
            }
            sHISA[bl * 2 * DD + c] = (bf16)(ldw1 * sum);
        }
        __syncthreads();

        // ========== Phase E: peer (two accumulated WMMA GEMMs) + prob ==========
        for (int j = wv; j < 24; j += NWAVES) {
            v8f acc;
#pragma unroll
            for (int i = 0; i < 8; ++i) acc[i] = 0.f;
            acc = tile_mm_pk(sBGA, 384, wpBGINT, 12, j, acc);
            acc = tile_mm_pk(sHISA, 128, wpHISINT, 4, j, acc);
            store_c_f32(sPEER, 384, j * 16, acc, sBC);
        }
        __syncthreads();
        if (tid < BLK_B) {
            int bl = tid, b = b0 + bl;
            float sum = b_pred[0];
            for (int k = 0; k < 384; ++k) sum += sPEER[bl * 384 + k] * W_pred[k];
            for (int k = 0; k < 128; ++k) sum += sV[bl * 128 + k] * W_pred[384 + k];
            for (int k = 0; k < 64;  ++k) sum += sH[bl * 64 + k]  * W_pred[512 + k];
            out[(long)b * TT + t] = sum;
        }

        // ========== Phase F: GRU input map + gates + state update ==========
        for (int i = tid; i < BLK_B * 640; i += NTHREADS) {
            int bl = i / 640, c = i % 640;
            float respf = (float)response[t * BB + (b0 + bl)];
            float val;
            if (c < 384)      val = sPEER[bl * 384 + c];
            else if (c < 512) val = sV[bl * 128 + (c - 384)] * respf;
            else              val = sV[bl * 128 + (c - 512)] * (1.f - respf);
            sACAT[bl * 640 + c] = (bf16)val;
        }
        for (int i = tid; i < BLK_B * DD; i += NTHREADS) sHB[i] = (bf16)sH[i];
        __syncthreads();
        for (int j = wv; j < 16; j += NWAVES) {       // gru_in -> bf16 directly (allhv dead)
            v8f acc;
#pragma unroll
            for (int i = 0; i < 8; ++i) acc[i] = 0.f;
            acc = tile_mm_pk(sACAT, 640, wpINMAP, 20, j, acc);
            store_c_bf16(sGRUB, 256, j * 16, acc, b_inmap);
        }
        __syncthreads();
        for (int j = wv; j < 24; j += NWAVES) {
            v8f acc;
#pragma unroll
            for (int i = 0; i < 8; ++i) acc[i] = 0.f;
            if (j < 12) {
                acc = tile_mm_pk(sGRUB, 256, wpIH, 8, j, acc);
                store_c_f32(sG1, 192, j * 16, acc, b_ih);
            } else {
                acc = tile_mm_pk(sHB, 64, wpHH, 2, j - 12, acc);
                store_c_f32(sG2, 192, (j - 12) * 16, acc, b_hh);
            }
        }
        __syncthreads();
        for (int i = tid; i < BLK_B * DD; i += NTHREADS) {
            int bl = i >> 6, d = i & 63;
            float xr = sG1[bl * 192 + d],        hr = sG2[bl * 192 + d];
            float xz = sG1[bl * 192 + 64 + d],   hz = sG2[bl * 192 + 64 + d];
            float xn = sG1[bl * 192 + 128 + d],  hn = sG2[bl * 192 + 128 + d];
            float r = 1.f / (1.f + __expf(-(xr + hr)));
            float z = 1.f / (1.f + __expf(-(xz + hz)));
            float n = tanhf(xn + r * hn);
            float nh = (1.f - z) * n + z * sH[i];
            sH[i] = nh;
            hbuf[((long)(b0 + bl) * (TT + 1) + (t + 1)) * DD + d] = nh;
        }
        for (int i = tid; i < BLK_B * 32; i += NTHREADS) {
            int bl = i >> 5, c4 = (i & 31) << 2;
            *(float4*)(vbuf + ((long)(b0 + bl) * (TT + 1) + (t + 1)) * (2 * DD) + c4) =
                *(const float4*)(sV + bl * 2 * DD + c4);
        }
        if (tid < BLK_B)
            sRB[tid * (TT + 1) + (t + 1)] = (float)response[t * BB + (b0 + tid)];
        __syncthreads();
    }
}

extern "C" void kernel_launch(void* const* d_in, const int* in_sizes, int n_in,
                              void* d_out, int out_size, void* d_ws, size_t ws_size,
                              hipStream_t stream) {
    const int*   prob_id      = (const int*)  d_in[0];
    const int*   skills       = (const int*)  d_in[1];
    const int*   response     = (const int*)  d_in[2];
    const int*   bg_index     = (const int*)  d_in[3];
    const int*   mem_prob_ids = (const int*)  d_in[4];
    const int*   mem_resp     = (const int*)  d_in[5];
    const int*   mem_concepts = (const int*)  d_in[6];
    const float* states_mem   = (const float*)d_in[7];
    const float* concept_emb  = (const float*)d_in[8];
    const float* prob_emb     = (const float*)d_in[9];
    const float* W_pred       = (const float*)d_in[10];
    const float* b_pred       = (const float*)d_in[11];
    const float* W_ih         = (const float*)d_in[12];
    const float* W_hh         = (const float*)d_in[13];
    const float* b_ih         = (const float*)d_in[14];
    const float* b_hh         = (const float*)d_in[15];
    const float* W_batch      = (const float*)d_in[16];
    const float* b_batch      = (const float*)d_in[17];
    const float* W_bg         = (const float*)d_in[18];
    const float* b_bgint      = (const float*)d_in[21];
    const float* W_bgint      = (const float*)d_in[20];
    const float* W_hisint     = (const float*)d_in[22];
    const float* b_hisint     = (const float*)d_in[23];
    const float* W_inmap      = (const float*)d_in[24];
    const float* b_inmap      = (const float*)d_in[25];
    const float* dir_w        = (const float*)d_in[26];
    const float* heads_map    = (const float*)d_in[27];

    bf16* wp = (bf16*)((char*)d_ws + WS_PACK_BYTE);
    // pack weights to bf16 fragment-native layout (runs before persistent kernel on same stream)
    pack_weight<<<dim3(96), dim3(256), 0, stream>>>(W_batch,  wp + OFFP_BATCH, 192, 64, 64, 1, 4, 12288L);
    pack_weight<<<dim3(96), dim3(256), 0, stream>>>(W_bg,     wp + OFFP_BG,     64, 192, 1, 64, 4, 12288L); // W_bg[h]^T
    pack_weight<<<dim3(96), dim3(256), 0, stream>>>(W_bgint,  wp + OFFP_BGINT, 384, 384, 384, 1, 1, 0L);
    pack_weight<<<dim3(96), dim3(256), 0, stream>>>(W_hisint, wp + OFFP_HISINT, 128, 384, 384, 1, 1, 0L);
    pack_weight<<<dim3(96), dim3(256), 0, stream>>>(W_inmap,  wp + OFFP_INMAP, 640, 256, 256, 1, 1, 0L);
    pack_weight<<<dim3(96), dim3(256), 0, stream>>>(W_ih,     wp + OFFP_IH,    256, 192, 192, 1, 1, 0L);
    pack_weight<<<dim3(96), dim3(256), 0, stream>>>(W_hh,     wp + OFFP_HH,     64, 192, 192, 1, 1, 0L);

    cokt_persistent<<<dim3(BB / BLK_B), dim3(NTHREADS), SMEM_TOTAL, stream>>>(
        prob_id, skills, response, bg_index, mem_prob_ids, mem_resp, mem_concepts,
        states_mem, concept_emb, prob_emb,
        W_pred, b_pred, b_ih, b_hh, b_batch,
        b_bgint, b_hisint, b_inmap, dir_w, heads_map,
        (float*)d_out, (float*)d_ws);
}
